// MatchAttention_13881334301213
// MI455X (gfx1250) — matrix operand order repaired and checked by reading, not compiled, not run on hardware
//
#include <hip/hip_runtime.h>
#include <cstdint>
#include <cstddef>

typedef __bf16 bf16_t;
typedef __bf16 v4bf  __attribute__((ext_vector_type(4)));
typedef __bf16 v8bf  __attribute__((ext_vector_type(8)));
typedef __bf16 v16bf __attribute__((ext_vector_type(16)));
typedef float  v8f   __attribute__((ext_vector_type(8)));

// Native conversions: clang lowers fptrunc/fpext for __bf16 on gfx1250
// (hardware cvt where available, correct round-to-nearest-even).
__device__ __forceinline__ bf16_t to_bf16(float f)      { return (bf16_t)f; }
__device__ __forceinline__ float  bf16_to_f32(bf16_t x) { return (float)x; }

__device__ __forceinline__ void store_out(float* p, float v)  { *p = v; }
__device__ __forceinline__ void store_out(bf16_t* p, float v) { *p = to_bf16(v); }

// ---- CDNA5 async global->LDS (ASYNCcnt-tracked DMA, no VGPR round trip) ----
typedef __attribute__((address_space(3))) void lds_void_t;

__device__ __forceinline__ unsigned lds_addr(void* p) {
    // generic -> LDS addrspace cast; AS(3) pointer value == LDS byte offset
    return (unsigned)(uintptr_t)(lds_void_t*)p;
}

__device__ __forceinline__ void async_load_b128(void* lds_dst, const void* gsrc) {
    asm volatile("global_load_async_to_lds_b128 %0, %1, off"
                 :: "v"(lds_addr(lds_dst)), "v"(gsrc)
                 : "memory");
}

template<int N>
__device__ __forceinline__ void wait_asynccnt() {
#if __has_builtin(__builtin_amdgcn_s_wait_asynccnt)
    __builtin_amdgcn_s_wait_asynccnt((unsigned short)N);
#else
    asm volatile("s_wait_asynccnt %0" :: "n"(N) : "memory");
#endif
}

// ---------------------------------------------------------------------------
// fp32 -> bf16 elementwise conversion (n must be a multiple of 4)
// ---------------------------------------------------------------------------
__global__ __launch_bounds__(256)
void cvt_f32_bf16(const float* __restrict__ in, bf16_t* __restrict__ out, int n) {
    int i = (blockIdx.x * 256 + threadIdx.x) * 4;
    if (i + 3 < n) {
        float4 v = *(const float4*)(in + i);
        v4bf o;
        o[0] = to_bf16(v.x); o[1] = to_bf16(v.y);
        o[2] = to_bf16(v.z); o[3] = to_bf16(v.w);
        *(v4bf*)(out + i) = o;   // single b64 store
    }
}

// ---------------------------------------------------------------------------
// Batched bf16 WMMA GEMM: C[b] = A[b] (MxK, row-major, lda=K) x B[b] + bias
//  B_KMAJOR=true : B source is [N][K] (K contiguous)  -> async DMA into LDS
//  B_KMAJOR=false: B source is [K][N] (N contiguous)  -> transpose via LDS (sync)
//  A_F32         : A source fp32, converted on the sync staging path;
//                  otherwise A is async-DMA'd.
// Block tile 128x128x32, 256 threads = 8 waves (2M x 4N wave grid, each wave
// a 64x32 sub-tile = 4x2 fragments), 2-deep LDS pipeline (32 KB of 320 KB).
// Software-pipelined: tile k+1 DMA issued before computing tile k.
// ---------------------------------------------------------------------------
template<bool A_F32, bool B_KMAJOR, bool BIAS, typename OutT>
__global__ __launch_bounds__(256)
void gemm_bf16_wmma(const void* __restrict__ Aptr,
                    const bf16_t* __restrict__ Bptr,
                    const float* __restrict__ biasPtr,
                    OutT* __restrict__ Cptr,
                    int M, int N, int K, int ldB,
                    long long strideA, long long strideB, long long strideC)
{
    constexpr int BM = 128, BN = 128, BK = 32;
    // async loads issued per thread per tile (A:2 unless fp32, B:2 if K-major)
    constexpr int NASYNC = (A_F32 ? 0 : 2) + (B_KMAJOR ? 2 : 0);

    __shared__ __align__(32) bf16_t As[2][BM * BK];
    __shared__ __align__(32) bf16_t Bs[2][BN * BK];

    const int tid  = threadIdx.x;
    const int wave = tid >> 5;
    const int lane = tid & 31;
    const int wm   = wave & 1;      // 0..1 -> 64 rows each
    const int wn   = wave >> 1;     // 0..3 -> 32 cols each
    const int kh   = lane >> 4;     // lane half (ISA fragment layouts)
    const int ln16 = lane & 15;

    const int m0 = blockIdx.y * BM;
    const int n0 = blockIdx.x * BN;

    const float*  Af = (const float*)Aptr  + (size_t)blockIdx.z * (size_t)strideA;
    const bf16_t* Ab = (const bf16_t*)Aptr + (size_t)blockIdx.z * (size_t)strideA;
    const bf16_t* Bg = Bptr                + (size_t)blockIdx.z * (size_t)strideB;
    OutT*         C  = Cptr                + (size_t)blockIdx.z * (size_t)strideC;

    // per-thread staging coordinates
    const int lr  = tid >> 1;          // 0..127 (row of A / col of B)
    const int ls  = (tid & 1) * 8;     // bf16 async segment (8 elems = 16 B)
    const int lfs = (tid & 1) * 16;    // fp32 path segment (16 elems)
    const int tkk = tid >> 3;          // transpose path: source k-row 0..31
    const int tcs = (tid & 7) * 16;    // transpose path: col segment

    auto load_tile = [&](int k0, int buf) {
        // ---- A tile (BM x BK) ----
        if constexpr (A_F32) {
            const float* src = Af + (size_t)(m0 + lr) * K + k0 + lfs;
            bf16_t* dst = &As[buf][lr * BK + lfs];
            #pragma unroll
            for (int q = 0; q < 4; ++q) {
                float4 v = *(const float4*)(src + q * 4);
                v4bf o;
                o[0] = to_bf16(v.x); o[1] = to_bf16(v.y);
                o[2] = to_bf16(v.z); o[3] = to_bf16(v.w);
                *(v4bf*)(dst + q * 4) = o;
            }
        } else {
            const bf16_t* src = Ab + (size_t)(m0 + lr) * K + k0 + ls;
            async_load_b128(&As[buf][lr * BK + ls],      src);
            async_load_b128(&As[buf][lr * BK + ls + 16], src + 16);
        }
        // ---- B tile (BN cols x BK k, stored [col][k]) ----
        if constexpr (B_KMAJOR) {
            const bf16_t* src = Bg + (size_t)(n0 + lr) * ldB + k0 + ls;
            async_load_b128(&Bs[buf][lr * BK + ls],      src);
            async_load_b128(&Bs[buf][lr * BK + ls + 16], src + 16);
        } else {
            v16bf v = *(const v16bf*)(Bg + (size_t)(k0 + tkk) * ldB + n0 + tcs);
            #pragma unroll
            for (int j = 0; j < 16; ++j)
                Bs[buf][(tcs + j) * BK + tkk] = v[j];   // transpose into [col][k]
        }
    };

    v8f acc[4][2] = {};

    auto compute = [&](int buf) {
        const bf16_t* AsB = As[buf];
        const bf16_t* BsB = Bs[buf];
        v16bf a[4], b[2];
        #pragma unroll
        for (int mt = 0; mt < 4; ++mt) {
            // A 16x32 frag: lane row = ln16; VGPR0-3: K=kh*8+0..7, VGPR4-7: K=16+kh*8+0..7
            const bf16_t* ap = &AsB[(wm * 64 + mt * 16 + ln16) * BK + kh * 8];
            v8bf h0 = *(const v8bf*)(ap);
            v8bf h1 = *(const v8bf*)(ap + 16);
            #pragma unroll
            for (int i = 0; i < 8; ++i) { a[mt][i] = h0[i]; a[mt][8 + i] = h1[i]; }
        }
        #pragma unroll
        for (int nt = 0; nt < 2; ++nt) {
            // B 32x16 frag: lane col = ln16; VGPR0-3: K=kh*8+0..7, VGPR4-7: K=16+kh*8+0..7
            const bf16_t* bp = &BsB[(wn * 32 + nt * 16 + ln16) * BK + kh * 8];
            v8bf g0 = *(const v8bf*)(bp);
            v8bf g1 = *(const v8bf*)(bp + 16);
            #pragma unroll
            for (int i = 0; i < 8; ++i) { b[nt][i] = g0[i]; b[nt][8 + i] = g1[i]; }
        }
        #pragma unroll
        for (int mt = 0; mt < 4; ++mt)
            #pragma unroll
            for (int nt = 0; nt < 2; ++nt)
                acc[mt][nt] = __builtin_amdgcn_wmma_f32_16x16x32_bf16(
                    false, a[mt], false, b[nt], (short)0, acc[mt][nt], false, false);
    };

    const int nk = K / BK;   // >= 2 for all our shapes

    load_tile(0, 0);
    for (int kt = 0; kt < nk - 1; ++kt) {
        load_tile((kt + 1) * BK, (kt + 1) & 1);   // prefetch next tile
        wait_asynccnt<NASYNC>();                  // previous tile's DMAs done
        __syncthreads();
        compute(kt & 1);
        __syncthreads();
    }
    wait_asynccnt<0>();
    __syncthreads();
    compute((nk - 1) & 1);

    // ---- epilogue: C/D layout: VGPR v -> row = v + 8*kh, col = ln16 ----
    #pragma unroll
    for (int mt = 0; mt < 4; ++mt) {
        #pragma unroll
        for (int nt = 0; nt < 2; ++nt) {
            const int gcol  = n0 + wn * 32 + nt * 16 + ln16;
            const int grow0 = m0 + wm * 64 + mt * 16 + kh * 8;
            float bias = 0.0f;
            if constexpr (BIAS) bias = biasPtr[gcol];
            #pragma unroll
            for (int v = 0; v < 8; ++v)
                store_out(C + (size_t)(grow0 + v) * N + gcol, acc[mt][nt][v] + bias);
        }
    }
}

// ---------------------------------------------------------------------------
// In-place row softmax on bf16 logits. One block per row of 2048 columns.
// ---------------------------------------------------------------------------
__global__ __launch_bounds__(256)
void softmax_inplace(bf16_t* __restrict__ attn, int ncols) {
    constexpr int VPT = 8;   // 2048 / 256
    bf16_t* p = attn + (size_t)blockIdx.x * ncols;
    const int tid = threadIdx.x;

    __shared__ float redm[8];
    __shared__ float reds[8];

    float x[VPT];
    float m = -1e30f;
    #pragma unroll
    for (int i = 0; i < VPT; ++i) {
        x[i] = bf16_to_f32(p[tid + i * 256]);
        m = fmaxf(m, x[i]);
    }
    #pragma unroll
    for (int o = 16; o > 0; o >>= 1) m = fmaxf(m, __shfl_xor(m, o, 32));
    if ((tid & 31) == 0) redm[tid >> 5] = m;
    __syncthreads();
    float bm = redm[0];
    #pragma unroll
    for (int w = 1; w < 8; ++w) bm = fmaxf(bm, redm[w]);

    float s = 0.0f;
    #pragma unroll
    for (int i = 0; i < VPT; ++i) { x[i] = __expf(x[i] - bm); s += x[i]; }
    #pragma unroll
    for (int o = 16; o > 0; o >>= 1) s += __shfl_xor(s, o, 32);
    if ((tid & 31) == 0) reds[tid >> 5] = s;
    __syncthreads();
    float bs = 0.0f;
    #pragma unroll
    for (int w = 0; w < 8; ++w) bs += reds[w];

    const float inv = 1.0f / bs;
    #pragma unroll
    for (int i = 0; i < VPT; ++i) p[tid + i * 256] = to_bf16(x[i] * inv);
}

// ---------------------------------------------------------------------------
extern "C" void kernel_launch(void* const* d_in, const int* in_sizes, int n_in,
                              void* d_out, int out_size, void* d_ws, size_t ws_size,
                              hipStream_t stream) {
    (void)in_sizes; (void)n_in; (void)out_size; (void)ws_size;
    constexpr int B = 4, L = 2048, D1 = 1024, D2 = 1024;

    const float* f1 = (const float*)d_in[0];   // [B,L,D1]
    const float* f2 = (const float*)d_in[1];   // [B,L,D2]
    const float* W1 = (const float*)d_in[2];   // [D2,D1]
    const float* b1 = (const float*)d_in[3];   // [D2]
    const float* W2 = (const float*)d_in[4];   // [D1,D2]
    const float* b2 = (const float*)d_in[5];   // [D1]

    float* out1 = (float*)d_out;                      // attended_1 [B,L,D2]
    float* out2 = out1 + (size_t)B * L * D2;          // attended_2 [B,L,D1]

    char* ws = (char*)d_ws;
    bf16_t* f2b  = (bf16_t*)ws;  ws += (size_t)B * L * D2 * 2;   // 16 MB
    bf16_t* W1b  = (bf16_t*)ws;  ws += (size_t)D2 * D1 * 2;      //  2 MB
    bf16_t* W2b  = (bf16_t*)ws;  ws += (size_t)D1 * D2 * 2;      //  2 MB
    bf16_t* t1   = (bf16_t*)ws;  ws += (size_t)B * L * D2 * 2;   // 16 MB
    bf16_t* attn = (bf16_t*)ws;  ws += (size_t)B * L * L * 2;    // 32 MB
    bf16_t* a2p  = (bf16_t*)ws;  ws += (size_t)B * L * D2 * 2;   // 16 MB

    // 0) fp32 -> bf16 conversions
    cvt_f32_bf16<<<dim3((unsigned)((size_t)B * L * D2 / 1024)), 256, 0, stream>>>(
        f2, f2b, B * L * D2);
    cvt_f32_bf16<<<dim3((unsigned)((size_t)D2 * D1 / 1024)), 256, 0, stream>>>(
        W1, W1b, D2 * D1);
    cvt_f32_bf16<<<dim3((unsigned)((size_t)D1 * D2 / 1024)), 256, 0, stream>>>(
        W2, W2b, D1 * D2);

    // 1) t1 = f1 @ W1^T + b1   (flattened M = B*L, N = D2, K = D1; W1 is [N][K])
    gemm_bf16_wmma<true, true, true, bf16_t>
        <<<dim3(D2 / 128, (B * L) / 128, 1), 256, 0, stream>>>(
        f1, W1b, b1, t1, B * L, D2, D1, /*ldB=*/D1, 0, 0, 0);

    // 2) sim logits = t1 @ f2^T  (per batch: M=L, N=L, K=D2; f2 is [N][K])
    gemm_bf16_wmma<false, true, false, bf16_t>
        <<<dim3(L / 128, L / 128, B), 256, 0, stream>>>(
        t1, f2b, nullptr, attn, L, L, D2, /*ldB=*/D2,
        (long long)L * D2, (long long)L * D2, (long long)L * L);

    // 3) row softmax (in place, bf16)
    softmax_inplace<<<dim3(B * L), 256, 0, stream>>>(attn, L);

    // 4) attended_1 = attn @ f2  (per batch: M=L, N=D2, K=L; f2 is [K][N])
    gemm_bf16_wmma<false, false, false, float>
        <<<dim3(D2 / 128, L / 128, B), 256, 0, stream>>>(
        attn, f2b, nullptr, out1, L, D2, L, /*ldB=*/D2,
        (long long)L * L, (long long)L * D2, (long long)L * D2);

    // 5) a2pre = attn @ t1  (per batch: M=L, N=D2, K=L; t1 is [K][N])
    gemm_bf16_wmma<false, false, false, bf16_t>
        <<<dim3(D2 / 128, L / 128, B), 256, 0, stream>>>(
        attn, t1, nullptr, a2p, L, D2, L, /*ldB=*/D2,
        (long long)L * L, (long long)L * D2, (long long)L * D2);

    // 6) attended_2 = a2pre @ W2^T + b2  (per batch: M=L, N=D1, K=D2; W2 is [N][K])
    gemm_bf16_wmma<false, true, true, float>
        <<<dim3(D1 / 128, L / 128, B), 256, 0, stream>>>(
        a2p, W2b, b2, out2, L, D1, D2, /*ldB=*/D2,
        (long long)L * D2, 0, (long long)L * D1);
}
